// MultiHeadAttention_16630113370583
// MI455X (gfx1250) — matrix operand order repaired and checked by reading, not compiled, not run on hardware
//
#include <hip/hip_runtime.h>
#include <hip/hip_bf16.h>

// ---------------------------------------------------------------------------
// MHA on gfx1250: bf16 WMMA GEMMs + flash-attention with mask-after-softmax
// correction. Round 3: tile staging now uses CDNA5 async global->LDS DMA
// (global_load_async_to_lds_b128 + s_wait_asynccnt) instead of VGPR bounce.
// ---------------------------------------------------------------------------

typedef __attribute__((ext_vector_type(16))) __bf16 v16bf;
typedef __attribute__((ext_vector_type(8)))  float  v8f;

union Op {                 // one WMMA A or B operand (16x32 / 32x16 bf16)
  v16bf v;
  uint4 q[2];
  unsigned short s[16];
};
union Acc {                // one 16x16 f32 C/D tile
  v8f v;
  float f[8];
};

__device__ __forceinline__ unsigned short f2bf(float f) {
  unsigned u = __float_as_uint(f);
  unsigned r = 0x7FFFu + ((u >> 16) & 1u);   // round-to-nearest-even
  return (unsigned short)((u + r) >> 16);
}

__device__ __forceinline__ v8f wmma_bf16(const Op& a, const Op& b, v8f c) {
  return __builtin_amdgcn_wmma_f32_16x16x32_bf16(
      false, a.v, false, b.v, (short)0, c, false, false);
}

// Async global->LDS 16B copy (VGLOBAL encoding, GV mode): VDST = LDS byte
// address (= low 32 bits of generic shared pointer), VADDR = 64-bit address.
// Tracked by ASYNCcnt; fenced with s_wait_asynccnt before the block barrier.
__device__ __forceinline__ void async_ld_b128(void* lds_dst, const void* gsrc) {
  unsigned lds = (unsigned)(unsigned long long)lds_dst;
  unsigned long long ga = (unsigned long long)gsrc;
  asm volatile("global_load_async_to_lds_b128 %0, %1, off"
               :: "v"(lds), "v"(ga));
}
__device__ __forceinline__ void wait_async0() {
  asm volatile("s_wait_asynccnt 0x0" ::: "memory");
}

static constexpr int S  = 2048;
static constexpr int D  = 1024;
static constexpr int NH = 16;
static constexpr int HD = 64;
static constexpr int M  = 4096;   // B*S

// LDS row paddings (elements): byte strides 144 / 80 are 16B-aligned and give
// bank-start sequences conflict-free across 16 lanes of b128 reads.
static constexpr int KP = 72;     // K tile rows of 64 -> pad 72
static constexpr int VP = 40;     // V/W tile rows of 32 -> pad 40
static constexpr int PP = 40;     // P staging rows of 32 -> pad 40

// -------------------------------- convert ----------------------------------
__global__ void cvt_bf16(const float* __restrict__ src,
                         unsigned short* __restrict__ dst, int n) {
  int i = blockIdx.x * blockDim.x + threadIdx.x;
  int stride = gridDim.x * blockDim.x;
  for (; i < n; i += stride) dst[i] = f2bf(src[i]);
}

// ---------------------------- mask bit-packing -----------------------------
// mask[S,S] int32 -> mpk[S][S/32] bits (bit j of word w = mask[q, w*32+j])
__global__ void pack_mask(const int* __restrict__ mask,
                          unsigned* __restrict__ mpk, int n) {
  int i = blockIdx.x * blockDim.x + threadIdx.x;
  int stride = gridDim.x * blockDim.x;
  for (; i < n; i += stride) {
    int q = i >> 6, w = i & 63;
    const int* p = mask + q * S + w * 32;
    unsigned bits = 0;
#pragma unroll
    for (int j = 0; j < 32; j++) bits |= (p[j] != 0 ? 1u : 0u) << j;
    mpk[i] = bits;
  }
}

// ------------------------------ GEMM  Y = A @ W^T ---------------------------
// A: [M,1024] bf16. W: [1024,1024] bf16 (B[k,n]=W[n,k]). Block: 8 waves share
// one 64-col W slab, async-staged in double-buffered LDS; each wave owns a
// 32x64 tile.
// mode 0: bf16 split-head [B,H,S,hd]; 1: bf16 [B,H,hd,S]; 2: f32 [M,1024]
__global__ __launch_bounds__(256) void gemm_bf16(
    const unsigned short* __restrict__ A,
    const unsigned short* __restrict__ W,
    void* __restrict__ dst, int mode) {
  __shared__ unsigned short ldsW[2][64 * VP];

  const int tid = threadIdx.x;
  const int lane = tid & 31;
  const int wave = tid >> 5;
  const int tn = blockIdx.x & 15;        // N/64 = 16
  const int mb = blockIdx.x >> 4;        // 16 m-blocks of 256 rows
  const int m0 = (mb * 8 + wave) * 32;
  const int n0 = tn * 64;
  const int g = lane >> 4, ln = lane & 15;
  const int K = 1024;

  const int wrow = tid >> 2, wch = tid & 3;   // 64 rows x 4 chunks of 8 elems
  auto stageW = [&](int k0, int buf) {
    async_ld_b128(&ldsW[buf][wrow * VP + wch * 8],
                  W + (n0 + wrow) * K + k0 + wch * 8);
  };

  Acc acc[2][4];
#pragma unroll
  for (int i = 0; i < 2; i++)
#pragma unroll
    for (int j = 0; j < 4; j++)
#pragma unroll
      for (int v = 0; v < 8; v++) acc[i][j].f[v] = 0.f;

  stageW(0, 0);
  wait_async0();
  __syncthreads();

  for (int it = 0; it < 32; ++it) {
    const int k0 = it * 32, buf = it & 1;
    if (it + 1 < 32) stageW(k0 + 32, buf ^ 1);

    Op a[2];
#pragma unroll
    for (int mi = 0; mi < 2; mi++) {
      const unsigned short* p = A + (m0 + mi * 16 + ln) * K + k0 + g * 8;
      a[mi].q[0] = *(const uint4*)p;
      a[mi].q[1] = *(const uint4*)(p + 16);
    }
    Op b[4];
#pragma unroll
    for (int ni = 0; ni < 4; ni++) {
      const unsigned short* p = &ldsW[buf][(ni * 16 + ln) * VP + g * 8];
      b[ni].q[0] = *(const uint4*)p;
      b[ni].q[1] = *(const uint4*)(p + 16);
    }
#pragma unroll
    for (int mi = 0; mi < 2; mi++)
#pragma unroll
      for (int ni = 0; ni < 4; ni++)
        acc[mi][ni].v = wmma_bf16(a[mi], b[ni], acc[mi][ni].v);

    wait_async0();
    __syncthreads();
  }

#pragma unroll
  for (int mi = 0; mi < 2; mi++)
#pragma unroll
    for (int ni = 0; ni < 4; ni++)
#pragma unroll
      for (int v = 0; v < 8; v++) {
        int m = m0 + mi * 16 + v + 8 * g;   // C layout: row = v + 8*g
        int n = n0 + ni * 16 + ln;          //           col = lane%16
        float val = acc[mi][ni].f[v];
        if (mode == 2) {
          ((float*)dst)[m * D + n] = val;
        } else {
          int bb = m >> 11, s = m & (S - 1);
          int h = n >> 6, d = n & (HD - 1);
          unsigned short bv = f2bf(val);
          if (mode == 0)
            ((unsigned short*)dst)[(((bb * NH) + h) * S + s) * HD + d] = bv;
          else
            ((unsigned short*)dst)[(((bb * NH) + h) * HD + d) * S + s] = bv;
        }
      }
}

// ------------------------- flash attention ---------------------------------
// Block = 8 waves sharing one (b,h); wave w handles 16 q-rows. Per 32-key
// step, K (32x64) and V^T (64x32) tiles are async-staged in double-buffered
// LDS.  out = (1/Z) sum_{mask=1} e^{s-m} v_k  - 1e-9 * sum_{mask=0} v_k
__global__ __launch_bounds__(256) void attn_kernel(
    const unsigned short* __restrict__ qh,
    const unsigned short* __restrict__ kh,
    const unsigned short* __restrict__ vT,
    const unsigned* __restrict__ mpk,
    unsigned short* __restrict__ attn_out) {
  __shared__ unsigned short ldsK[2][32 * KP];
  __shared__ unsigned short ldsV[2][64 * VP];
  __shared__ unsigned short ldsP[8][16 * PP];
  __shared__ unsigned short ldsM[8][16 * PP];

  const int tid = threadIdx.x;
  const int lane = tid & 31;
  const int wave = tid >> 5;
  const int bh = blockIdx.x >> 4;          // 32 (b,h) pairs
  const int qblk = blockIdx.x & 15;        // 16 q-blocks of 128 rows
  const int b = bh >> 4, h = bh & 15;
  const int q0 = qblk * 128 + wave * 16;
  const int g = lane >> 4, ln = lane & 15;

  const unsigned short* qb = qh + (size_t)bh * S * HD;
  const unsigned short* kb = kh + (size_t)bh * S * HD;
  const unsigned short* vb = vT + (size_t)bh * HD * S;
  unsigned short* myP = ldsP[wave];
  unsigned short* myM = ldsM[wave];

  const int krow = tid >> 3, kch = tid & 7;   // 32 rows x 8 chunks
  const int vrow = tid >> 2, vch = tid & 3;   // 64 rows x 4 chunks
  auto stageKV = [&](int k0, int buf) {
    async_ld_b128(&ldsK[buf][krow * KP + kch * 8],
                  kb + (k0 + krow) * HD + kch * 8);
    async_ld_b128(&ldsV[buf][vrow * VP + vch * 8],
                  vb + vrow * S + k0 + vch * 8);
  };

  // preload Q operands: A[m,k] = qh[q0+m, dh*32 + k], k = 8g+s (s<8), 16+8g+s-8 (s>=8)
  Op aq[2];
#pragma unroll
  for (int dh = 0; dh < 2; dh++) {
    const unsigned short* p = qb + (q0 + ln) * HD + dh * 32 + g * 8;
    aq[dh].q[0] = *(const uint4*)p;
    aq[dh].q[1] = *(const uint4*)(p + 16);
  }

  Acc oacc[4], macc[4];
#pragma unroll
  for (int ni = 0; ni < 4; ni++)
#pragma unroll
    for (int v = 0; v < 8; v++) { oacc[ni].f[v] = 0.f; macc[ni].f[v] = 0.f; }

  float mrun[8], lrun[8];
#pragma unroll
  for (int v = 0; v < 8; v++) { mrun[v] = -1e30f; lrun[v] = 0.f; }

  stageKV(0, 0);
  wait_async0();
  __syncthreads();

  for (int it = 0; it < S / 32; ++it) {
    const int k0 = it * 32, buf = it & 1;
    if (it + 1 < S / 32) stageKV(k0 + 32, buf ^ 1);

    // ---- scores: 2 key-tiles of 16, K=64 over d (2 chained WMMAs each)
    Acc sc[2];
#pragma unroll
    for (int t = 0; t < 2; t++) {
      Op kb0, kb1;
      const unsigned short* p0 = &ldsK[buf][(t * 16 + ln) * KP + g * 8];
      kb0.q[0] = *(const uint4*)p0;
      kb0.q[1] = *(const uint4*)(p0 + 16);
      const unsigned short* p1 = p0 + 32;
      kb1.q[0] = *(const uint4*)p1;
      kb1.q[1] = *(const uint4*)(p1 + 16);
      v8f z = {};
      sc[t].v = wmma_bf16(aq[0], kb0, z);
      sc[t].v = wmma_bf16(aq[1], kb1, sc[t].v);
    }
    const float rs = 0.125f;  // 1/sqrt(hd)
#pragma unroll
    for (int t = 0; t < 2; t++)
#pragma unroll
      for (int v = 0; v < 8; v++) sc[t].f[v] *= rs;

    // ---- online softmax stats (row m = v+8g lives in one VGPR / 16 lanes)
    float nm[8], ls[8];
#pragma unroll
    for (int v = 0; v < 8; v++) {
      float mx = fmaxf(sc[0].f[v], sc[1].f[v]);
#pragma unroll
      for (int xm = 1; xm < 16; xm <<= 1) mx = fmaxf(mx, __shfl_xor(mx, xm, 16));
      nm[v] = fmaxf(mrun[v], mx);
      float e0 = __expf(sc[0].f[v] - nm[v]);
      float e1 = __expf(sc[1].f[v] - nm[v]);
      sc[0].f[v] = e0; sc[1].f[v] = e1;
      float sum = e0 + e1;                       // Z over ALL keys (pre-mask)
#pragma unroll
      for (int xm = 1; xm < 16; xm <<= 1) sum += __shfl_xor(sum, xm, 16);
      ls[v] = sum;
    }

    // ---- packed mask (post-softmax semantics), stage P / (1-mask) to LDS
#pragma unroll
    for (int v = 0; v < 8; v++) {
      int row = v + 8 * g;
      unsigned word = mpk[(q0 + row) * (S / 32) + it];
#pragma unroll
      for (int t = 0; t < 2; t++) {
        unsigned mv = (word >> (t * 16 + ln)) & 1u;
        float pv = mv ? sc[t].f[v] : 0.f;
        myP[row * PP + t * 16 + ln] = f2bf(pv);
        myM[row * PP + t * 16 + ln] =
            mv ? (unsigned short)0 : (unsigned short)0x3F80;  // bf16 1.0
      }
    }

    // ---- rescale running state
#pragma unroll
    for (int v = 0; v < 8; v++) {
      float alpha = __expf(mrun[v] - nm[v]);
      lrun[v] = lrun[v] * alpha + ls[v];
      mrun[v] = nm[v];
#pragma unroll
      for (int ni = 0; ni < 4; ni++) oacc[ni].f[v] *= alpha;
    }

    // ---- reload P / maskP in A-operand layout from LDS (same wave, in-order)
    Op pa, pm;
    {
      const unsigned short* pp = myP + ln * PP + g * 8;
      pa.q[0] = *(const uint4*)pp;
      pa.q[1] = *(const uint4*)(pp + 16);
      const unsigned short* pq = myM + ln * PP + g * 8;
      pm.q[0] = *(const uint4*)pq;
      pm.q[1] = *(const uint4*)(pq + 16);
    }

    // ---- P@V and maskP@V from LDS V tile
#pragma unroll
    for (int ni = 0; ni < 4; ni++) {
      Op vv;
      const unsigned short* vp = &ldsV[buf][(ni * 16 + ln) * VP + g * 8];
      vv.q[0] = *(const uint4*)vp;
      vv.q[1] = *(const uint4*)(vp + 16);
      oacc[ni].v = wmma_bf16(pa, vv, oacc[ni].v);
      macc[ni].v = wmma_bf16(pm, vv, macc[ni].v);
    }

    wait_async0();
    __syncthreads();
  }

  // ---- epilogue: normalize + masked correction, write [B,S,D] bf16
#pragma unroll
  for (int ni = 0; ni < 4; ni++)
#pragma unroll
    for (int v = 0; v < 8; v++) {
      int s = q0 + v + 8 * g;
      int d = h * HD + ni * 16 + ln;
      float val = oacc[ni].f[v] / lrun[v] - 1e-9f * macc[ni].f[v];
      attn_out[((size_t)b * S + s) * D + d] = f2bf(val);
    }
}

// ------------------------------- launcher ----------------------------------
extern "C" void kernel_launch(void* const* d_in, const int* in_sizes, int n_in,
                              void* d_out, int out_size, void* d_ws,
                              size_t ws_size, hipStream_t stream) {
  const float* q    = (const float*)d_in[0];
  const float* k    = (const float*)d_in[1];
  const float* v    = (const float*)d_in[2];
  const int*   mask = (const int*)d_in[3];
  const float* wq   = (const float*)d_in[4];
  const float* wk   = (const float*)d_in[5];
  const float* wv   = (const float*)d_in[6];
  const float* wout = (const float*)d_in[7];
  float* out = (float*)d_out;

  const int nBig = M * D;     // 4096*1024
  const int nW   = D * D;     // 1024*1024
  const int nMpk = S * (S / 32);

  unsigned short* Xq  = (unsigned short*)d_ws;
  unsigned short* Xk  = Xq  + nBig;
  unsigned short* Xv  = Xk  + nBig;
  unsigned short* Wqb = Xv  + nBig;
  unsigned short* Wkb = Wqb + nW;
  unsigned short* Wvb = Wkb + nW;
  unsigned short* Wob = Wvb + nW;
  unsigned short* qh  = Wob + nW;
  unsigned short* kh  = qh  + nBig;
  unsigned short* vT  = kh  + nBig;
  unsigned short* ao  = vT  + nBig;
  unsigned*       mpk = (unsigned*)(ao + nBig);   // ~64.5 MB total

  cvt_bf16<<<512, 256, 0, stream>>>(q, Xq, nBig);
  cvt_bf16<<<512, 256, 0, stream>>>(k, Xk, nBig);
  cvt_bf16<<<512, 256, 0, stream>>>(v, Xv, nBig);
  cvt_bf16<<<256, 256, 0, stream>>>(wq, Wqb, nW);
  cvt_bf16<<<256, 256, 0, stream>>>(wk, Wkb, nW);
  cvt_bf16<<<256, 256, 0, stream>>>(wv, Wvb, nW);
  cvt_bf16<<<256, 256, 0, stream>>>(wout, Wob, nW);
  pack_mask<<<512, 256, 0, stream>>>(mask, mpk, nMpk);

  gemm_bf16<<<256, 256, 0, stream>>>(Xq, Wqb, (void*)qh, 0);
  gemm_bf16<<<256, 256, 0, stream>>>(Xk, Wkb, (void*)kh, 0);
  gemm_bf16<<<256, 256, 0, stream>>>(Xv, Wvb, (void*)vT, 1);

  attn_kernel<<<512, 256, 0, stream>>>(qh, kh, vT, mpk, ao);

  gemm_bf16<<<256, 256, 0, stream>>>(ao, Wob, (void*)out, 2);
}
